// LabelWiseAttention_89223650607792
// MI455X (gfx1250) — matrix operand order repaired and checked by reading, not compiled, not run on hardware
//
#include <hip/hip_runtime.h>

#define B_   8
#define S_   512
#define D_   1024
#define C_   14268
#define CTILE 64
#define NTHREADS 128
#define LSTRIDE (D_ + 8)           // bf16 elements per LDS row (pad vs bank period)
#define SBLOCK 128                 // s positions resident in LDS per block
#define NCHUNK (SBLOCK / 16)       // 8 WMMA m-tiles per block
#define NBLOCKS (S_ / SBLOCK)      // 4
#define NKSTEP (D_ / 32)           // 32 WMMA k-steps over D (fully unrolled)

typedef __bf16 bf16;
typedef __attribute__((ext_vector_type(16))) __bf16 v16bf;
typedef __attribute__((ext_vector_type(8)))  __bf16 v8bf;
typedef __attribute__((ext_vector_type(4)))  __bf16 v4bf;
typedef __attribute__((ext_vector_type(8)))  float  v8f;
typedef __attribute__((ext_vector_type(4)))  float  v4f;

extern __shared__ char smem_raw[];

// Load 16 consecutive fp32 (64B) and convert to one bf16 WMMA B-fragment.
__device__ __forceinline__ v16bf load_bfrag(const float* __restrict__ p) {
    v4f f0 = *(const v4f*)(p + 0);
    v4f f1 = *(const v4f*)(p + 4);
    v4f f2 = *(const v4f*)(p + 8);
    v4f f3 = *(const v4f*)(p + 12);
    v16bf r;
    #pragma unroll
    for (int i = 0; i < 4; ++i) {
        r[i]      = (__bf16)f0[i];
        r[4 + i]  = (__bf16)f1[i];
        r[8 + i]  = (__bf16)f2[i];
        r[12 + i] = (__bf16)f3[i];
    }
    return r;
}

__global__ __launch_bounds__(NTHREADS)
void lwa_fused_kernel(const float* __restrict__ x,
                      const float* __restrict__ Wa,
                      const float* __restrict__ Wo,
                      const float* __restrict__ bo,
                      float* __restrict__ out)
{
    bf16* sX = (bf16*)smem_raw;                  // [SBLOCK][LSTRIDE] bf16

    const int tid  = threadIdx.x;
    const int b    = blockIdx.y;
    const int c0   = blockIdx.x * CTILE;
    const int wave = tid >> 5;
    const int lane = tid & 31;
    const int ln   = lane & 15;                  // label column owned by this lane
    const int half = lane >> 4;

    // This lane's label row (clamped: garbage columns are store-guarded, stays finite)
    const int c    = c0 + wave * 16 + ln;
    const int crow = (c < C_) ? c : 0;
    // B-frag per k-step: elements k0 + half*16 + {0..15}  (32x16 bf16 B layout)
    const float* waRow = Wa + (size_t)crow * D_ + half * 16;
    const float* woRow = Wo + (size_t)crow * D_ + half * 16;
    // A-frag per k-step: lanes 0-15 hold K {0..15}, lanes 16-31 {16..31}
    const int kA0 = half * 16;

    // Fixed per-chunk LDS base pointers: all k offsets become DS immediate
    // offsets -> no VALU address math in the WMMA loop (no WMMA->VALU WAR NOPs).
    const bf16* xptr[NCHUNK];
    #pragma unroll
    for (int ch = 0; ch < NCHUNK; ++ch)
        xptr[ch] = sX + (ch * 16 + ln) * LSTRIDE + kA0;

    // online-softmax running state (per lane: one label, its half of each s-chunk)
    float m = -__builtin_inff();
    float l = 0.f;
    float acc = 0.f;

    for (int blk = 0; blk < NBLOCKS; ++blk) {
        __syncthreads();  // previous block's compute done before sX overwrite

        // ---- cooperative load + fp32->bf16 convert of x s-block into LDS ----
        const float* xsrc = x + ((size_t)b * S_ + blk * SBLOCK) * D_;
        for (int i = tid; i < SBLOCK * (D_ / 4); i += NTHREADS) {
            const int r = i >> 8;                // D_/4 == 256
            const int q = i & 255;
            v4f fx = *(const v4f*)(xsrc + r * D_ + q * 4);     // global_load_b128
            v4bf bx = {(__bf16)fx[0], (__bf16)fx[1], (__bf16)fx[2], (__bf16)fx[3]};
            *(v4bf*)(sX + r * LSTRIDE + q * 4) = bx;           // ds_store_b64
        }
        __syncthreads();

        // prefetch hint for next s-block (128 thr x 16 x 256B = 512KB = full block)
        if (blk + 1 < NBLOCKS) {
            const char* nx = (const char*)(xsrc + (size_t)SBLOCK * D_);
            #pragma unroll
            for (int pf = 0; pf < 16; ++pf)
                __builtin_prefetch(nx + ((size_t)tid + (size_t)pf * NTHREADS) * 256, 0, 1);
        }

        // ---- accumulators: a and p tiles for 8 s-chunks, resident in VGPRs ----
        v8f ca[NCHUNK], cp[NCHUNK];
        #pragma unroll
        for (int ch = 0; ch < NCHUNK; ++ch) {
            ca[ch] = (v8f){0.f, 0.f, 0.f, 0.f, 0.f, 0.f, 0.f, 0.f};
            cp[ch] = (v8f){0.f, 0.f, 0.f, 0.f, 0.f, 0.f, 0.f, 0.f};
        }

        // ---- fully unrolled K loop. Per k-step: this label's Wa/Wo frags come
        //      from L2 (fp32->bf16 cvt co-executes with XDL WMMA); all 16
        //      A-frag ds_load_b128 are batched BEFORE the 16 WMMAs so the
        //      scheduler can use incremental s_wait_dscnt and overlap the next
        //      k-step's loads with this one's WMMAs. ----
        #pragma unroll
        for (int t = 0; t < NKSTEP; ++t) {
            const v16bf bw = load_bfrag(waRow + t * 32);
            const v16bf bv = load_bfrag(woRow + t * 32);

            v8bf alo[NCHUNK], ahi[NCHUNK];
            #pragma unroll
            for (int ch = 0; ch < NCHUNK; ++ch) {
                alo[ch] = *(const v8bf*)(xptr[ch] + t * 32);        // ds_load_b128 imm
                ahi[ch] = *(const v8bf*)(xptr[ch] + t * 32 + 8);    // ds_load_b128 imm
            }
            #pragma unroll
            for (int ch = 0; ch < NCHUNK; ++ch) {
                v16bf af = __builtin_shufflevector(alo[ch], ahi[ch],
                    0, 1, 2, 3, 4, 5, 6, 7, 8, 9, 10, 11, 12, 13, 14, 15);
                ca[ch] = __builtin_amdgcn_wmma_f32_16x16x32_bf16(
                    false, af, false, bw, (short)0, ca[ch], false, false);
                cp[ch] = __builtin_amdgcn_wmma_f32_16x16x32_bf16(
                    false, af, false, bv, (short)0, cp[ch], false, false);
            }
        }

        // ---- online softmax update per s-chunk (in-lane, 8 s-positions) ----
        #pragma unroll
        for (int ch = 0; ch < NCHUNK; ++ch) {
            float tmax = ca[ch][0];
            #pragma unroll
            for (int i = 1; i < 8; ++i) tmax = fmaxf(tmax, ca[ch][i]);
            const float mn = fmaxf(m, tmax);
            const float rescale = __expf(m - mn);   // first update: exp(-inf) = 0
            l *= rescale;
            acc *= rescale;
            #pragma unroll
            for (int i = 0; i < 8; ++i) {
                const float w = __expf(ca[ch][i] - mn);
                l += w;
                acc = fmaf(w, cp[ch][i], acc);
            }
            m = mn;
        }
    }

    // ---- merge the two lane-halves (s%16 in 0..7 vs 8..15) ----
    const float m2 = __shfl_xor(m, 16, 32);
    const float l2 = __shfl_xor(l, 16, 32);
    const float a2 = __shfl_xor(acc, 16, 32);
    const float M  = fmaxf(m, m2);
    const float w1 = __expf(m - M);
    const float w2 = __expf(m2 - M);
    const float L  = l * w1 + l2 * w2;
    const float A  = acc * w1 + a2 * w2;

    if (half == 0 && c < C_) {
        const float logit = A / L + bo[c];
        out[(size_t)b * C_ + c] = 1.f / (1.f + __expf(-logit));
    }
}

extern "C" void kernel_launch(void* const* d_in, const int* in_sizes, int n_in,
                              void* d_out, int out_size, void* d_ws, size_t ws_size,
                              hipStream_t stream) {
    const float* x  = (const float*)d_in[0];
    const float* Wa = (const float*)d_in[1];
    const float* Wo = (const float*)d_in[2];
    const float* bo = (const float*)d_in[3];
    float* out = (float*)d_out;

    const size_t shmem = (size_t)SBLOCK * LSTRIDE * sizeof(bf16);  // ~258 KB of 320 KB WGP LDS
    (void)hipFuncSetAttribute((const void*)lwa_fused_kernel,
                              hipFuncAttributeMaxDynamicSharedMemorySize, (int)shmem);

    dim3 grid((C_ + CTILE - 1) / CTILE, B_);  // 223 x 8 = 1784 workgroups
    lwa_fused_kernel<<<grid, NTHREADS, shmem, stream>>>(x, Wa, Wo, bo, out);
}
